// BERTGNN_68066641707093
// MI455X (gfx1250) — matrix operand order repaired and checked by reading, not compiled, not run on hardware
//
#include <hip/hip_runtime.h>

// ---------------- problem constants ----------------
#define NNODE 20000
#define NEDGE 240000
#define ETOT  (NEDGE + NNODE)   // edges + self loops
#define DIN   768
#define DD    256
#define LLAY  3
#define HHEAD 4
#define DHEAD 64
#define TTAG  9
#define NETYPE 8
#define BN_EPS 1e-5f

// ---------------- types for WMMA ----------------
typedef __bf16 v16bf __attribute__((ext_vector_type(16)));
typedef __bf16 v8bf  __attribute__((ext_vector_type(8)));
typedef float  v8f   __attribute__((ext_vector_type(8)));
typedef unsigned short us8v __attribute__((ext_vector_type(8)));

// ---------------- helpers ----------------
__device__ __forceinline__ unsigned short f2bf(float x) {
  unsigned u = __float_as_uint(x);
  u += 0x7fffu + ((u >> 16) & 1u);        // round-to-nearest-even
  return (unsigned short)(u >> 16);
}
__device__ __forceinline__ unsigned fmono(float f) {   // order-preserving f32->u32
  unsigned u = __float_as_uint(f);
  return (u & 0x80000000u) ? ~u : (u | 0x80000000u);
}
__device__ __forceinline__ float monof(unsigned u) {
  return (u & 0x80000000u) ? __uint_as_float(u & 0x7fffffffu) : __uint_as_float(~u);
}
__device__ __forceinline__ float gelu_f(float x) {
  return 0.5f * x * (1.0f + erff(x * 0.70710678118654752f));
}
__device__ __forceinline__ void edge_get(const int* ei, const int* et, int e,
                                         int& s, int& d, int& ty) {
  if (e < NEDGE) { s = ei[e]; d = ei[NEDGE + e]; ty = et[e]; }
  else           { s = e - NEDGE; d = s; ty = NETYPE; }   // self loop, type NE
}

// ---------------- tiny utility kernels ----------------
__global__ void bg_zero(float* p, long n) {
  long i = (long)blockIdx.x * blockDim.x + threadIdx.x;
  if (i < n) p[i] = 0.0f;
}
__global__ void bg_cast_bf16(const float* s, unsigned short* d, long n) {
  long i = (long)blockIdx.x * blockDim.x + threadIdx.x;
  if (i < n) d[i] = f2bf(s[i]);
}
__global__ void bg_deg(const int* ei, float* deg) {
  int e = blockIdx.x * blockDim.x + threadIdx.x;
  if (e >= ETOT) return;
  int s = (e < NEDGE) ? ei[e] : (e - NEDGE);
  atomicAdd(deg + s, 1.0f);
}

// Pack a K x 256 row-major f32 weight into WMMA-B bf16 fragments.
// frag element i of lane L holds B[kt*32 + 16*(L>>4) + i][nt*16 + (L&15)]
__global__ void bg_pack_b(const float* W, unsigned short* Bp, int K, int N) {
  int t = blockIdx.x * blockDim.x + threadIdx.x;
  int nkt = K >> 5, nnt = N >> 4;
  if (t >= nkt * nnt * 32) return;
  int lane = t & 31;
  int nt = (t >> 5) % nnt;
  int kt = (t >> 5) / nnt;
  int c = lane & 15, hh = lane >> 4;
  unsigned short* out = Bp + (size_t)t * 16;
#pragma unroll
  for (int i = 0; i < 16; i++) {
    int k = kt * 32 + 16 * hh + i;
    out[i] = f2bf(W[(size_t)k * N + nt * 16 + c]);
  }
}

// ---------------- WMMA GEMM: C[M x 256] = act(A[M x K](bf16) @ Bp + bias (+Cadd)) ---------
#define ACT_NONE   0
#define ACT_GELU   1
#define ACT_BNRELU 2

__global__ __launch_bounds__(256) void bg_gemm(
    const unsigned short* __restrict__ A, int K,
    const unsigned short* __restrict__ Bp,
    const float* __restrict__ bias,
    const float* __restrict__ bng, const float* __restrict__ bnb,
    const float* __restrict__ Cadd,
    float* __restrict__ Cf, unsigned short* __restrict__ Cbf,
    int Mtiles, int act)
{
  int wave = threadIdx.x >> 5;
  int lane = threadIdx.x & 31;
  int mt = blockIdx.x * 8 + wave;
  if (mt >= Mtiles) return;
  int nt0 = blockIdx.y * 8;              // N = 256 -> 16 col tiles, 8 per wave
  int c = lane & 15, h = lane >> 4;

  v8f acc[8];
#pragma unroll
  for (int i = 0; i < 8; i++)
#pragma unroll
    for (int j = 0; j < 8; j++) acc[i][j] = 0.0f;

  const unsigned short* Arow = A + (size_t)(mt * 16 + c) * K;
  int nkt = K >> 5;
  for (int kt = 0; kt < nkt; kt++) {
    // A fragment: K = 16h..16h+15  (two contiguous 16B loads)
    const unsigned short* p = Arow + kt * 32 + 16 * h;
    us8v lo = *(const us8v*)(p);
    us8v hi = *(const us8v*)(p + 8);
    v16bf a = __builtin_shufflevector(__builtin_bit_cast(v8bf, lo),
                                      __builtin_bit_cast(v8bf, hi),
                                      0, 1, 2, 3, 4, 5, 6, 7, 8, 9, 10, 11, 12, 13, 14, 15);
    const unsigned short* bpk = Bp + ((size_t)(kt * 16 + nt0) * 32 + lane) * 16;
#pragma unroll
    for (int nt = 0; nt < 8; nt++) {
      v16bf b = *(const v16bf*)(bpk + (size_t)nt * 32 * 16);
      acc[nt] = __builtin_amdgcn_wmma_f32_16x16x32_bf16(
          false, a, false, b, (short)0, acc[nt], false, false);
    }
  }

  float bnr = rsqrtf(1.0f + BN_EPS);
#pragma unroll
  for (int nt = 0; nt < 8; nt++) {
    int col = (nt0 + nt) * 16 + c;
    float bs = bias ? bias[col] : 0.0f;
    float g = 0.0f, bb = 0.0f;
    if (act == ACT_BNRELU) { g = bng[col] * bnr; bb = bnb[col]; }
#pragma unroll
    for (int j = 0; j < 8; j++) {
      int row = mt * 16 + 8 * h + j;
      float v = acc[nt][j] + bs;
      if (Cadd) v += Cadd[(size_t)row * DD + col];
      if (act == ACT_GELU)        v = gelu_f(v);
      else if (act == ACT_BNRELU) { v = v * g + bb; v = v < 0.0f ? 0.0f : v; }
      if (Cf)  Cf[(size_t)row * DD + col] = v;
      if (Cbf) Cbf[(size_t)row * DD + col] = f2bf(v);
    }
  }
}

// ---------------- edge-type tables (one-hot strength reduction) ----------------
// ke[t] = relu(bn(We1[t]+be1)) @ We2 + be2  then  @ Wk[D:], @ Wm[D:]
__global__ __launch_bounds__(256) void bg_etable(
    const float* We1l, const float* be1l, const float* g1l, const float* bb1l,
    const float* We2l, const float* be2l,
    const float* WkEl, const float* WmEl,
    float* ketab, float* metab)
{
  __shared__ float tmp[TTAG][DD];
  __shared__ float et[TTAG][DD];
  int cth = threadIdx.x;
  float s = rsqrtf(1.0f + BN_EPS);
  for (int t = 0; t < TTAG; t++) {
    float x = We1l[t * DD + cth] + be1l[cth];
    x = x * (g1l[cth] * s) + bb1l[cth];
    tmp[t][cth] = x > 0.0f ? x : 0.0f;
  }
  __syncthreads();
  for (int t = 0; t < TTAG; t++) {
    float a = be2l[cth];
    for (int k = 0; k < DD; k++) a += tmp[t][k] * We2l[(size_t)k * DD + cth];
    et[t][cth] = a;
  }
  __syncthreads();
  for (int t = 0; t < TTAG; t++) {
    float a = 0.0f, b = 0.0f;
    for (int k = 0; k < DD; k++) {
      float ev = et[t][k];
      a += ev * WkEl[(size_t)k * DD + cth];
      b += ev * WmEl[(size_t)k * DD + cth];
    }
    ketab[t * DD + cth] = a;
    metab[t * DD + cth] = b;
  }
}

// ---------------- edge attention passes ----------------
__global__ void bg_edge_score(const int* ei, const int* et,
                              const float* __restrict__ Qn, const float* __restrict__ Kxn,
                              const float* __restrict__ ketab,
                              float* sbuf, unsigned* smax)
{
  int gid = blockIdx.x * blockDim.x + threadIdx.x;
  if (gid >= ETOT * HHEAD) return;
  int e = gid >> 2, h = gid & 3;
  int s, d, ty; edge_get(ei, et, e, s, d, ty);
  const float* q  = Qn  + (size_t)s * DD + h * DHEAD;
  const float* kx = Kxn + (size_t)d * DD + h * DHEAD;
  const float* ke = ketab + ty * DD + h * DHEAD;
  float acc = 0.0f;
#pragma unroll 8
  for (int i = 0; i < DHEAD; i++) acc += q[i] * (kx[i] + ke[i]);
  acc *= 0.125f;                       // 1/sqrt(64)
  sbuf[gid] = acc;
  atomicMax(smax + s * HHEAD + h, fmono(acc));
}

__global__ void bg_edge_exp(const int* ei, const int* et,
                            float* sbuf, const unsigned* smax, float* den)
{
  int gid = blockIdx.x * blockDim.x + threadIdx.x;
  if (gid >= ETOT * HHEAD) return;
  int e = gid >> 2, h = gid & 3;
  int s, d, ty; edge_get(ei, et, e, s, d, ty);
  float ex = expf(sbuf[gid] - monof(smax[s * HHEAD + h]));
  sbuf[gid] = ex;
  atomicAdd(den + s * HHEAD + h, ex);
}

__global__ void bg_edge_aggr(const int* ei, const int* et,
                             const float* sbuf, const float* den, const float* deg,
                             const float* __restrict__ Mxn, const float* __restrict__ metab,
                             float* aggr)
{
  int gid = blockIdx.x * blockDim.x + threadIdx.x;
  if (gid >= ETOT * HHEAD) return;
  int e = gid >> 2, h = gid & 3;
  int s, d, ty; edge_get(ei, et, e, s, d, ty);
  float alpha = sbuf[gid] / (den[s * HHEAD + h] + 1e-16f) * deg[s];
  const float* mx = Mxn + (size_t)s * DD + h * DHEAD;
  const float* me = metab + ty * DD + h * DHEAD;
  float* ag = aggr + (size_t)d * DD + h * DHEAD;
#pragma unroll 8
  for (int i = 0; i < DHEAD; i++) atomicAdd(ag + i, (mx[i] + me[i]) * alpha);
}

// ---------------- emissions + Viterbi ----------------
__global__ void bg_emis(const float* __restrict__ hidden, const float* __restrict__ Wt,
                        const float* __restrict__ bt, float* em)
{
  int n = blockIdx.x * blockDim.x + threadIdx.x;
  if (n >= NNODE) return;
  float o[TTAG];
#pragma unroll
  for (int t = 0; t < TTAG; t++) o[t] = bt[t];
  const float* hr = hidden + (size_t)n * DD;
  for (int k = 0; k < DD; k++) {
    float hv = hr[k];
#pragma unroll
    for (int t = 0; t < TTAG; t++) o[t] += hv * Wt[k * TTAG + t];
  }
#pragma unroll
  for (int t = 0; t < TTAG; t++) em[(size_t)n * TTAG + t] = o[t];
}

__global__ void bg_viterbi(const float* __restrict__ em, const float* cs, const float* ce,
                           const float* tr, unsigned char* bp, float* out)
{
  __shared__ float sc[TTAG];
  __shared__ float ns[TTAG];
  int t = threadIdx.x;
  if (t < TTAG) sc[t] = cs[t] + em[t];
  __syncthreads();
  for (int step = 1; step < NNODE; step++) {
    if (t < TTAG) {
      float best = -3.4e38f; int bi = 0;
#pragma unroll
      for (int i = 0; i < TTAG; i++) {
        float v = sc[i] + tr[i * TTAG + t];
        if (v > best) { best = v; bi = i; }
      }
      bp[(size_t)(step - 1) * TTAG + t] = (unsigned char)bi;
      ns[t] = best + em[(size_t)step * TTAG + t];
    }
    __syncthreads();
    if (t < TTAG) sc[t] = ns[t];
    __syncthreads();
  }
  if (t == 0) {
    float best = -3.4e38f; int bi = 0;
    for (int i = 0; i < TTAG; i++) {
      float v = sc[i] + ce[i];
      if (v > best) { best = v; bi = i; }
    }
    int cur = bi;
    out[NNODE - 1] = (float)cur;
    for (int step = NNODE - 2; step >= 0; step--) {
      cur = bp[(size_t)step * TTAG + cur];
      out[step] = (float)cur;
    }
  }
}

// ---------------- host orchestration ----------------
extern "C" void kernel_launch(void* const* d_in, const int* in_sizes, int n_in,
                              void* d_out, int out_size, void* d_ws, size_t ws_size,
                              hipStream_t stream)
{
  (void)in_sizes; (void)n_in; (void)out_size; (void)ws_size;
  const float* node_emb = (const float*)d_in[0];
  const int*   edge_index = (const int*)d_in[1];
  const int*   edge_type  = (const int*)d_in[2];
  const float* W_lm2gnn = (const float*)d_in[4];
  const float* b_lm2gnn = (const float*)d_in[5];
  const float* We1 = (const float*)d_in[6];
  const float* be1 = (const float*)d_in[7];
  const float* g1  = (const float*)d_in[8];
  const float* bb1 = (const float*)d_in[9];
  const float* We2 = (const float*)d_in[10];
  const float* be2 = (const float*)d_in[11];
  const float* Wq  = (const float*)d_in[12];
  const float* bq  = (const float*)d_in[13];
  const float* Wk  = (const float*)d_in[14];
  const float* bk  = (const float*)d_in[15];
  const float* Wm  = (const float*)d_in[16];
  const float* bm  = (const float*)d_in[17];
  const float* Wp1 = (const float*)d_in[18];
  const float* bp1 = (const float*)d_in[19];
  const float* g2  = (const float*)d_in[20];
  const float* bb2 = (const float*)d_in[21];
  const float* Wp2 = (const float*)d_in[22];
  const float* bp2 = (const float*)d_in[23];
  const float* Wo  = (const float*)d_in[24];
  const float* bo  = (const float*)d_in[25];
  const float* Wc  = (const float*)d_in[26];
  const float* bc  = (const float*)d_in[27];
  const float* Wt  = (const float*)d_in[28];
  const float* bt  = (const float*)d_in[29];
  const float* crf_start = (const float*)d_in[30];
  const float* crf_end   = (const float*)d_in[31];
  const float* crf_trans = (const float*)d_in[32];

  // ---- carve workspace ----
  char* wsb = (char*)d_ws;
  size_t off = 0;
  auto carve = [&](size_t bytes) -> char* {
    char* p = wsb + off;
    off += (bytes + 255) & ~(size_t)255;
    return p;
  };
  unsigned short* node_bf = (unsigned short*)carve((size_t)NNODE * DIN * 2);
  unsigned short* H0bf    = (unsigned short*)carve((size_t)NNODE * DD * 2);
  unsigned short* Xbf     = (unsigned short*)carve((size_t)NNODE * DD * 2);
  unsigned short* P1bf    = (unsigned short*)carve((size_t)NNODE * DD * 2);
  unsigned short* aggrbf  = (unsigned short*)carve((size_t)NNODE * DD * 2);
  float* aggr = (float*)carve((size_t)NNODE * DD * 4);
  float* Qn   = (float*)carve((size_t)NNODE * DD * 4);
  float* Kxn  = (float*)carve((size_t)NNODE * DD * 4);
  float* Mxn  = (float*)carve((size_t)NNODE * DD * 4);
  float* sbuf = (float*)carve((size_t)ETOT * HHEAD * 4);
  unsigned* smax = (unsigned*)carve((size_t)NNODE * HHEAD * 4);
  float* den  = (float*)carve((size_t)NNODE * HHEAD * 4);
  float* deg  = (float*)carve((size_t)NNODE * 4);
  float* ketab = (float*)carve((size_t)TTAG * DD * 4);
  float* metab = (float*)carve((size_t)TTAG * DD * 4);
  unsigned short* pw0  = (unsigned short*)carve((size_t)DIN * DD * 2);
  unsigned short* pwQ  = (unsigned short*)carve((size_t)DD * DD * 2);
  unsigned short* pwK  = (unsigned short*)carve((size_t)DD * DD * 2);
  unsigned short* pwM  = (unsigned short*)carve((size_t)DD * DD * 2);
  unsigned short* pwP1 = (unsigned short*)carve((size_t)DD * DD * 2);
  unsigned short* pwP2 = (unsigned short*)carve((size_t)DD * DD * 2);
  float* em = (float*)carve((size_t)NNODE * TTAG * 4);
  unsigned char* bp = (unsigned char*)carve((size_t)(NNODE - 1) * TTAG);

  auto cdiv = [](long a, long b) { return (int)((a + b - 1) / b); };
  const int Mtiles = NNODE / 16;                 // 1250, exact
  dim3 gN(cdiv(Mtiles, 8), 2);                   // node-level GEMM grid (N=256)
  const int packT = (DD / 32) * (DD / 16) * 32;  // 4096
  const int EHT = ETOT * HHEAD;

  // H0 = gelu(node_emb @ W_lm2gnn + b)
  bg_cast_bf16<<<cdiv((long)NNODE * DIN, 256), 256, 0, stream>>>(node_emb, node_bf, (long)NNODE * DIN);
  bg_pack_b<<<cdiv((DIN / 32) * (DD / 16) * 32, 256), 256, 0, stream>>>(W_lm2gnn, pw0, DIN, DD);
  bg_gemm<<<gN, 256, 0, stream>>>(node_bf, DIN, pw0, b_lm2gnn, nullptr, nullptr,
                                  nullptr, nullptr, H0bf, Mtiles, ACT_GELU);

  // src out-degree (static across layers)
  bg_zero<<<cdiv(NNODE, 256), 256, 0, stream>>>(deg, NNODE);
  bg_deg<<<cdiv(ETOT, 256), 256, 0, stream>>>(edge_index, deg);

  const unsigned short* Xcur = H0bf;
  for (int l = 0; l < LLAY; l++) {
    const float* WkE = Wk + (size_t)l * 2 * DD * DD + (size_t)DD * DD;
    const float* WmE = Wm + (size_t)l * 2 * DD * DD + (size_t)DD * DD;
    bg_etable<<<1, 256, 0, stream>>>(We1 + (size_t)l * (NETYPE + 1) * DD, be1 + l * DD,
                                     g1 + l * DD, bb1 + l * DD,
                                     We2 + (size_t)l * DD * DD, be2 + l * DD,
                                     WkE, WmE, ketab, metab);
    bg_pack_b<<<cdiv(packT, 256), 256, 0, stream>>>(Wq + (size_t)l * DD * DD, pwQ, DD, DD);
    bg_pack_b<<<cdiv(packT, 256), 256, 0, stream>>>(Wk + (size_t)l * 2 * DD * DD, pwK, DD, DD);
    bg_pack_b<<<cdiv(packT, 256), 256, 0, stream>>>(Wm + (size_t)l * 2 * DD * DD, pwM, DD, DD);
    bg_pack_b<<<cdiv(packT, 256), 256, 0, stream>>>(Wp1 + (size_t)l * DD * DD, pwP1, DD, DD);
    bg_pack_b<<<cdiv(packT, 256), 256, 0, stream>>>(Wp2 + (size_t)l * DD * DD, pwP2, DD, DD);

    // node-level projections (gather folded into edge kernels)
    bg_gemm<<<gN, 256, 0, stream>>>(Xcur, DD, pwQ, bq + l * DD, nullptr, nullptr,
                                    nullptr, Qn, nullptr, Mtiles, ACT_NONE);
    bg_gemm<<<gN, 256, 0, stream>>>(Xcur, DD, pwK, bk + l * DD, nullptr, nullptr,
                                    nullptr, Kxn, nullptr, Mtiles, ACT_NONE);
    bg_gemm<<<gN, 256, 0, stream>>>(Xcur, DD, pwM, bm + l * DD, nullptr, nullptr,
                                    nullptr, Mxn, nullptr, Mtiles, ACT_NONE);

    bg_zero<<<cdiv(NNODE * HHEAD, 256), 256, 0, stream>>>((float*)smax, NNODE * HHEAD);
    bg_zero<<<cdiv(NNODE * HHEAD, 256), 256, 0, stream>>>(den, NNODE * HHEAD);
    bg_zero<<<cdiv((long)NNODE * DD, 256), 256, 0, stream>>>(aggr, (long)NNODE * DD);

    bg_edge_score<<<cdiv(EHT, 256), 256, 0, stream>>>(edge_index, edge_type, Qn, Kxn, ketab, sbuf, smax);
    bg_edge_exp<<<cdiv(EHT, 256), 256, 0, stream>>>(edge_index, edge_type, sbuf, smax, den);
    bg_edge_aggr<<<cdiv(EHT, 256), 256, 0, stream>>>(edge_index, edge_type, sbuf, den, deg, Mxn, metab, aggr);

    // X = gelu(relu(bn(aggr@Wp1+bp1)) @ Wp2 + bp2)
    bg_cast_bf16<<<cdiv((long)NNODE * DD, 256), 256, 0, stream>>>(aggr, aggrbf, (long)NNODE * DD);
    bg_gemm<<<gN, 256, 0, stream>>>(aggrbf, DD, pwP1, bp1 + l * DD, g2 + l * DD, bb2 + l * DD,
                                    nullptr, nullptr, P1bf, Mtiles, ACT_BNRELU);
    bg_gemm<<<gN, 256, 0, stream>>>(P1bf, DD, pwP2, bp2 + l * DD, nullptr, nullptr,
                                    nullptr, nullptr, Xbf, Mtiles, ACT_GELU);
    Xcur = Xbf;
  }

  // hidden = gelu(H0@Wo + bo + X@Wc + bc); reuse Qn as T1, Kxn as hidden
  float* T1 = Qn;
  float* hidden = Kxn;
  bg_pack_b<<<cdiv(packT, 256), 256, 0, stream>>>(Wo, pwQ, DD, DD);
  bg_pack_b<<<cdiv(packT, 256), 256, 0, stream>>>(Wc, pwK, DD, DD);
  bg_gemm<<<gN, 256, 0, stream>>>(H0bf, DD, pwQ, bo, nullptr, nullptr,
                                  nullptr, T1, nullptr, Mtiles, ACT_NONE);
  bg_gemm<<<gN, 256, 0, stream>>>(Xbf, DD, pwK, bc, nullptr, nullptr,
                                  T1, hidden, nullptr, Mtiles, ACT_GELU);

  bg_emis<<<cdiv(NNODE, 128), 128, 0, stream>>>(hidden, Wt, bt, em);
  bg_viterbi<<<1, 32, 0, stream>>>(em, crf_start, crf_end, crf_trans, bp, (float*)d_out);
}
